// AttentionDecoderRNN_2559800508963
// MI455X (gfx1250) — matrix operand order repaired and checked by reading, not compiled, not run on hardware
//
#include <hip/hip_runtime.h>
#include <hip/hip_bf16.h>

// Problem constants (from reference)
#define VSZ 32000
#define ESZ 2048
#define HSZ 2048
#define SSZ 64
#define BSZ 128
#define NEGBIG -1e10f

typedef __attribute__((ext_vector_type(16))) __bf16    v16bf;
typedef __attribute__((ext_vector_type(8)))  float     v8f;
typedef __attribute__((ext_vector_type(8)))  unsigned  v8u;

#if __has_builtin(__builtin_amdgcn_global_load_async_to_lds_b128) && \
    __has_builtin(__builtin_amdgcn_s_wait_asynccnt)
#define HAVE_ASYNC_LDS 1
// Pointee type per hipcc diagnostic: int __attribute__((vector_size(16)))
typedef int async_v4i __attribute__((vector_size(16)));
#else
#define HAVE_ASYNC_LDS 0
#endif

// Pack two fp32 into one dword of two bf16 (RNE). Prefer the native
// v_cvt_pk_bf16_f32; fall back to compiler bf16 conversion (same instruction
// via fptrunc) rather than integer bit-twiddling.
__device__ __forceinline__ unsigned pack2bf(float f0, float f1) {
#if __has_builtin(__builtin_amdgcn_cvt_pk_bf16_f32)
    typedef __attribute__((ext_vector_type(2))) __bf16 v2bf;
    v2bf p = __builtin_amdgcn_cvt_pk_bf16_f32(f0, f1);
    return __builtin_bit_cast(unsigned, p);
#else
    typedef __attribute__((ext_vector_type(2))) __bf16 v2bf;
    v2bf p;
    p[0] = (__bf16)f0;
    p[1] = (__bf16)f1;
    return __builtin_bit_cast(unsigned, p);
#endif
}

// A fragment: 16x32 bf16 tile of row-major fp32 A (lda = K), tile origin (m0, kk).
// ISA 7.12.2: lane m = lane&15; half-wave h=lane>>4 selects K runs 8h..8h+7 (VGPR 0-3)
// and 16+8h..16+8h+7 (VGPR 4-7). Per lane these are two contiguous 8-float runs
// -> four b128 loads.
__device__ __forceinline__ v16bf load_frag_A(const float* __restrict__ A, int lda,
                                             int m0, int kk, int lane) {
    const int m = m0 + (lane & 15);
    const int h = lane >> 4;
    const float* p0 = A + (size_t)m * (size_t)lda + kk + 8 * h;  // K = 8h .. 8h+7
    const float4 a0 = *(const float4*)(p0);
    const float4 a1 = *(const float4*)(p0 + 4);
    const float4 b0 = *(const float4*)(p0 + 16);                 // K = 16+8h ..
    const float4 b1 = *(const float4*)(p0 + 20);
    v8u pk;
    pk[0] = pack2bf(a0.x, a0.y); pk[1] = pack2bf(a0.z, a0.w);
    pk[2] = pack2bf(a1.x, a1.y); pk[3] = pack2bf(a1.z, a1.w);
    pk[4] = pack2bf(b0.x, b0.y); pk[5] = pack2bf(b0.z, b0.w);
    pk[6] = pack2bf(b1.x, b1.y); pk[7] = pack2bf(b1.z, b1.w);
    return __builtin_bit_cast(v16bf, pk);
}

// B fragment: 32x16 bf16 tile == W^T tile where W is (N,K) row-major fp32 with row
// stride ldw. lane n = lane&15; half-wave h selects K = 8h..8h+7 and 16+8h..16+8h+7:
// two contiguous 8-float runs -> four b128 loads.
__device__ __forceinline__ v16bf load_frag_B(const float* __restrict__ W, size_t ldw,
                                             int n0, int kk, int lane) {
    const int n = n0 + (lane & 15);
    const int h = lane >> 4;
    const float* p = W + (size_t)n * ldw + kk + 8 * h;
    const float4 a0 = *(const float4*)(p);
    const float4 a1 = *(const float4*)(p + 4);
    const float4 a2 = *(const float4*)(p + 16);
    const float4 a3 = *(const float4*)(p + 20);
    v8u pk;
    pk[0] = pack2bf(a0.x, a0.y); pk[1] = pack2bf(a0.z, a0.w);
    pk[2] = pack2bf(a1.x, a1.y); pk[3] = pack2bf(a1.z, a1.w);
    pk[4] = pack2bf(a2.x, a2.y); pk[5] = pack2bf(a2.z, a2.w);
    pk[6] = pack2bf(a3.x, a3.y); pk[7] = pack2bf(a3.z, a3.w);
    return __builtin_bit_cast(v16bf, pk);
}

__device__ __forceinline__ void gemm_accum(v8f& acc,
                                           const float* __restrict__ A, int K,
                                           const float* __restrict__ W, size_t ldw,
                                           int m0, int n0, int lane) {
    const float* wrow = W + (size_t)(n0 + (lane & 15)) * ldw;
#pragma unroll 2
    for (int kk = 0; kk < K; kk += 32) {
        // Unconditional prefetch of the next weight K-chunk (global_prefetch_b8);
        // speculative prefetch past the row end is silently dropped.
        __builtin_prefetch(wrow + kk + 64, 0, 1);
        v16bf a = load_frag_A(A, K, m0, kk, lane);
        v16bf b = load_frag_B(W, ldw, n0, kk, lane);
        acc = __builtin_amdgcn_wmma_f32_16x16x32_bf16(
            /*neg_a=*/false, a, /*neg_b=*/false, b,
            /*c_mod=*/(short)0, acc, /*reuse_a=*/false, /*reuse_b=*/false);
    }
}

// C[M=128, N] = act( A0 @ W0part^T + A1 @ W1part^T + bias0 + bias1 )
// Block = 256 threads = 8 waves; wave w owns m-tile w (M=128 exactly); grid.x = N/16.
__global__ __launch_bounds__(256) void gemm2_wmma_kernel(
    const float* __restrict__ A0, int K0, const float* __restrict__ W0, int ldw0,
    const float* __restrict__ A1, int K1, const float* __restrict__ W1, int ldw1,
    const float* __restrict__ bias0, const float* __restrict__ bias1,
    float* __restrict__ C, int N, int act) {
    const int lane = threadIdx.x & 31;
    const int wave = threadIdx.x >> 5;
    const int m0 = wave * 16;
    const int n0 = blockIdx.x * 16;

    v8f acc = {};
    gemm_accum(acc, A0, K0, W0, (size_t)ldw0, m0, n0, lane);
    if (A1) gemm_accum(acc, A1, K1, W1, (size_t)ldw1, m0, n0, lane);

    const int n = n0 + (lane & 15);
    const int hh = lane >> 4;
    float bv = 0.f;
    if (bias0) bv += bias0[n];
    if (bias1) bv += bias1[n];
#pragma unroll
    for (int r = 0; r < 8; ++r) {
        const int m = m0 + r + 8 * hh;
        float v = acc[r] + bv;
        if (act == 1) v = tanhf(v);
        C[(size_t)m * (size_t)N + n] = v;
    }
}

// X0[b] = [ embedding[dec_input[b]] | context[b] ]  -> (B, E+H)
__global__ __launch_bounds__(256) void embed_concat_kernel(
    const int* __restrict__ dec_input, const float* __restrict__ context,
    const float* __restrict__ embedding, float* __restrict__ X0) {
    const int b = blockIdx.x;
    const int tok = dec_input[b];  // dec_input is (B,1)
    const float* er = embedding + (size_t)tok * ESZ;
    float* xr = X0 + (size_t)b * (ESZ + HSZ);
    for (int i = threadIdx.x; i < ESZ; i += 256) xr[i] = er[i];
    for (int i = threadIdx.x; i < HSZ; i += 256) xr[ESZ + i] = context[(size_t)b * HSZ + i];
}

// gates G = (B, 4H) with order [i, f, g, o]; c2 = sig(f)*c + sig(i)*tanh(g); h2 = sig(o)*tanh(c2)
__global__ __launch_bounds__(256) void lstm_cell_kernel(
    const float* __restrict__ G, const float* __restrict__ c_prev,
    float* __restrict__ h_out, float* __restrict__ c_out) {
    const int idx = blockIdx.x * 256 + threadIdx.x;  // 0..B*H-1
    const int b = idx / HSZ, h = idx % HSZ;
    const float* g = G + (size_t)b * 4 * HSZ;
    const float i_ = g[h];
    const float f_ = g[HSZ + h];
    const float gg = g[2 * HSZ + h];
    const float o_ = g[3 * HSZ + h];
    const float si = 1.f / (1.f + __expf(-i_));
    const float sf = 1.f / (1.f + __expf(-f_));
    const float so = 1.f / (1.f + __expf(-o_));
    const float c2 = sf * c_prev[idx] + si * tanhf(gg);
    c_out[idx] = c2;
    h_out[idx] = so * tanhf(c2);
}

// One block per batch element b: scores[s] = enc[b,s,:]·x[b,:], mask, softmax over s,
// attn -> (S,B); ctx[b,:] = sum_s attn[s]*enc[b,s,:]. enc (67MB) fits in the 192MB L2,
// so the second pass over enc hits L2.
__global__ __launch_bounds__(256) void attention_kernel(
    const float* __restrict__ enc, const float* __restrict__ x,
    const int* __restrict__ src_len, float* __restrict__ attn_out,
    float* __restrict__ ctx) {
    __shared__ float xs[HSZ];
    __shared__ float sc[SSZ];
    const int b = blockIdx.x;

#if HAVE_ASYNC_LDS
    {
        // Stage x[b,:] (8KB) into LDS via the async DMA path (ASYNCcnt-tracked).
        const float* gsrc = x + (size_t)b * HSZ;
        for (int i = threadIdx.x; i < HSZ / 4; i += 256) {
            __builtin_amdgcn_global_load_async_to_lds_b128(
                (async_v4i*)(gsrc + 4 * i),
                (async_v4i*)&xs[4 * i], 0, 0);
        }
        __builtin_amdgcn_s_wait_asynccnt(0);
    }
#else
    for (int h = threadIdx.x; h < HSZ; h += 256) xs[h] = x[(size_t)b * HSZ + h];
#endif
    __syncthreads();

    const int wave = threadIdx.x >> 5, lane = threadIdx.x & 31;
    const float* eb = enc + (size_t)b * SSZ * HSZ;
    for (int s = wave; s < SSZ; s += 8) {
        const float* row = eb + (size_t)s * HSZ;
        float p = 0.f;
        for (int h = lane; h < HSZ; h += 32) p += row[h] * xs[h];
#pragma unroll
        for (int off = 16; off > 0; off >>= 1) p += __shfl_down(p, off, 32);
        if (lane == 0) sc[s] = p;
    }
    __syncthreads();

    const int len = src_len[b];
    if (threadIdx.x == 0) {  // S=64: serial softmax is negligible
        float mx = -3.0e38f;
        for (int s = 0; s < SSZ; ++s) {
            float m = (s < len) ? sc[s] : 0.f;  // seq_mask * score
            if (m == 0.f) m = NEGBIG;           // where(masked==0, NEG, masked)
            sc[s] = m;
            mx = fmaxf(mx, m);
        }
        float sum = 0.f;
        for (int s = 0; s < SSZ; ++s) { float e = __expf(sc[s] - mx); sc[s] = e; sum += e; }
        const float inv = 1.f / sum;
        for (int s = 0; s < SSZ; ++s) sc[s] *= inv;
    }
    __syncthreads();

    if (threadIdx.x < SSZ) attn_out[(size_t)threadIdx.x * BSZ + b] = sc[threadIdx.x];

    for (int h = threadIdx.x; h < HSZ; h += 256) {
        float acc = 0.f;
#pragma unroll 4
        for (int s = 0; s < SSZ; ++s) acc += sc[s] * eb[(size_t)s * HSZ + h];
        ctx[(size_t)b * HSZ + h] = acc;
    }
}

// per-row log-softmax over V=32000
__global__ __launch_bounds__(256) void logsoftmax_kernel(
    const float* __restrict__ logits, float* __restrict__ outp) {
    __shared__ float buf[256];
    const int b = blockIdx.x;
    const float* row = logits + (size_t)b * VSZ;

    float mx = -3.0e38f;
    for (int i = threadIdx.x; i < VSZ; i += 256) mx = fmaxf(mx, row[i]);
    buf[threadIdx.x] = mx;
    __syncthreads();
    for (int s = 128; s > 0; s >>= 1) {
        if (threadIdx.x < s) buf[threadIdx.x] = fmaxf(buf[threadIdx.x], buf[threadIdx.x + s]);
        __syncthreads();
    }
    mx = buf[0];
    __syncthreads();

    float sum = 0.f;
    for (int i = threadIdx.x; i < VSZ; i += 256) sum += __expf(row[i] - mx);
    buf[threadIdx.x] = sum;
    __syncthreads();
    for (int s = 128; s > 0; s >>= 1) {
        if (threadIdx.x < s) buf[threadIdx.x] += buf[threadIdx.x + s];
        __syncthreads();
    }
    const float lse = mx + __logf(buf[0]);

    for (int i = threadIdx.x; i < VSZ; i += 256) outp[(size_t)b * VSZ + i] = row[i] - lse;
}

extern "C" void kernel_launch(void* const* d_in, const int* in_sizes, int n_in,
                              void* d_out, int out_size, void* d_ws, size_t ws_size,
                              hipStream_t stream) {
    (void)in_sizes; (void)n_in; (void)out_size; (void)ws_size;
    // inputs in setup_inputs() order
    const int*   dec_input = (const int*)  d_in[0];
    const float* context   = (const float*)d_in[1];
    const float* prev_h    = (const float*)d_in[2];   // (2,B,H)
    const float* prev_c    = (const float*)d_in[3];   // (2,B,H)
    const float* enc       = (const float*)d_in[4];   // (B,S,H)
    const int*   src_len   = (const int*)  d_in[5];
    const float* embedding = (const float*)d_in[6];
    const float* W_ih0     = (const float*)d_in[7];   // (4H, E+H)
    const float* W_hh0     = (const float*)d_in[8];   // (4H, H)
    const float* b_ih0     = (const float*)d_in[9];
    const float* b_hh0     = (const float*)d_in[10];
    const float* W_ih1     = (const float*)d_in[11];  // (4H, H)
    const float* W_hh1     = (const float*)d_in[12];  // (4H, H)
    const float* b_ih1     = (const float*)d_in[13];
    const float* b_hh1     = (const float*)d_in[14];
    const float* att_l1_w  = (const float*)d_in[15];  // (H, H)
    const float* att_l2_w  = (const float*)d_in[16];  // (H, 2H)
    const float* fc_w      = (const float*)d_in[17];  // (V, H)
    const float* fc_b      = (const float*)d_in[18];

    float* out = (float*)d_out;
    // output tuple layout: out_vocab(B*V), out(B*H), hiddens(2*B*H), cs(2*B*H), attn(S*B)
    const size_t O_OUT  = (size_t)BSZ * VSZ;
    const size_t O_HID  = O_OUT + (size_t)BSZ * HSZ;
    const size_t O_CS   = O_HID + 2ull * BSZ * HSZ;
    const size_t O_ATTN = O_CS + 2ull * BSZ * HSZ;
    float* h0o = out + O_HID;
    float* h1o = out + O_HID + (size_t)BSZ * HSZ;
    float* c0o = out + O_CS;
    float* c1o = out + O_CS + (size_t)BSZ * HSZ;

    // workspace layout (floats)
    float* ws    = (float*)d_ws;
    float* X0    = ws;                                  // B*(E+H)
    float* G     = X0 + (size_t)BSZ * (ESZ + HSZ);      // B*4H
    float* XATT  = G + (size_t)BSZ * 4 * HSZ;           // B*H
    float* CTX   = XATT + (size_t)BSZ * HSZ;            // B*H
    float* LOGIT = CTX + (size_t)BSZ * HSZ;             // B*V

    const dim3 blk(256);

    // 1) embedding gather + concat with context -> X0 (B, E+H)
    embed_concat_kernel<<<BSZ, blk, 0, stream>>>(dec_input, context, embedding, X0);

    // 2) LSTM0 gates: G = X0 @ W_ih0^T + h0_prev @ W_hh0^T + b_ih0 + b_hh0   (B, 4H)
    gemm2_wmma_kernel<<<(4 * HSZ) / 16, blk, 0, stream>>>(
        X0, ESZ + HSZ, W_ih0, ESZ + HSZ,
        prev_h, HSZ, W_hh0, HSZ,
        b_ih0, b_hh0, G, 4 * HSZ, 0);
    lstm_cell_kernel<<<(BSZ * HSZ) / 256, blk, 0, stream>>>(G, prev_c, h0o, c0o);

    // 3) LSTM1 gates: G = h0 @ W_ih1^T + h1_prev @ W_hh1^T + biases
    gemm2_wmma_kernel<<<(4 * HSZ) / 16, blk, 0, stream>>>(
        h0o, HSZ, W_ih1, HSZ,
        prev_h + (size_t)BSZ * HSZ, HSZ, W_hh1, HSZ,
        b_ih1, b_hh1, G, 4 * HSZ, 0);
    lstm_cell_kernel<<<(BSZ * HSZ) / 256, blk, 0, stream>>>(
        G, prev_c + (size_t)BSZ * HSZ, h1o, c1o);

    // 4) x = h1 @ att_l1_w^T   (B, H)
    gemm2_wmma_kernel<<<HSZ / 16, blk, 0, stream>>>(
        h1o, HSZ, att_l1_w, HSZ,
        nullptr, 0, nullptr, 0,
        nullptr, nullptr, XATT, HSZ, 0);

    // 5) masked softmax attention: attn (S,B) -> d_out; ctx (B,H) -> ws
    attention_kernel<<<BSZ, blk, 0, stream>>>(enc, XATT, src_len, out + O_ATTN, CTX);

    // 6) out = tanh([ctx | h1] @ att_l2_w^T): split att_l2_w (H, 2H) along K
    gemm2_wmma_kernel<<<HSZ / 16, blk, 0, stream>>>(
        CTX, HSZ, att_l2_w, 2 * HSZ,
        h1o, HSZ, att_l2_w + HSZ, 2 * HSZ,
        nullptr, nullptr, out + O_OUT, HSZ, 1);

    // 7) logits = out @ fc_w^T + fc_b   (B, V)
    gemm2_wmma_kernel<<<VSZ / 16, blk, 0, stream>>>(
        out + O_OUT, HSZ, fc_w, HSZ,
        nullptr, 0, nullptr, 0,
        fc_b, nullptr, LOGIT, VSZ, 0);

    // 8) out_vocab = log_softmax(logits)
    logsoftmax_kernel<<<BSZ, blk, 0, stream>>>(LOGIT, out);
}
